// Encoder_LSTM_57372173140121
// MI455X (gfx1250) — matrix operand order repaired and checked by reading, not compiled, not run on hardware
//
#include <hip/hip_runtime.h>
#include <hip/hip_bf16.h>

typedef unsigned short u16;
typedef __attribute__((ext_vector_type(4)))  u16    u16x4;
typedef __attribute__((ext_vector_type(8)))  u16    u16x8;
typedef __attribute__((ext_vector_type(16))) __bf16 v16bf;
typedef __attribute__((ext_vector_type(8)))  float  v8f;

#define HDIM 512
#define KDIM 512
#define BM   128
#define BK   32
#define KT   (KDIM / BK)   // 16 K-steps
#define NGATE 3            // gates i, g, o; f is dead because c_prev == 0

union BFrag { u16x8 h[2]; v16bf v; };

__device__ __forceinline__ u16 f2bf(float f) {
  unsigned u = __float_as_uint(f);
  u += 0x7FFFu + ((u >> 16) & 1u);          // round to nearest even
  return (u16)(u >> 16);
}
__device__ __forceinline__ float bf2f(u16 h) {
  return __uint_as_float(((unsigned)h) << 16);
}
__device__ __forceinline__ BFrag ldsFrag(const u16* p) {
  BFrag f;
  f.h[0] = *(const u16x8*)(p);
  f.h[1] = *(const u16x8*)(p + 8);
  return f;
}
__device__ __forceinline__ float sigmoidf_(float x) {
  return 1.0f / (1.0f + __expf(-x));
}
__device__ __forceinline__ float tanhf_(float x) {
  x = fminf(fmaxf(x, -15.0f), 15.0f);
  float e = __expf(2.0f * x);
  return (e - 1.0f) / (e + 1.0f);
}

// ---- ragged-gather index construction -------------------------------------
__global__ void scan_kernel(const int* __restrict__ bs, int* __restrict__ cum, int T) {
  if (threadIdx.x == 0 && blockIdx.x == 0) {
    int acc = 0;
    for (int i = 0; i < T; ++i) { cum[i] = acc; acc += bs[i]; }
    cum[T] = acc;
  }
}

__global__ void rowmap_kernel(const int* __restrict__ bs, const int* __restrict__ cum,
                              int* __restrict__ rowmap, int Nrows, int T) {
  int r = blockIdx.x * blockDim.x + threadIdx.x;
  if (r >= Nrows) return;
  int lo = 0, hi = T - 1;
  while (lo < hi) {                          // largest i with cum[i] <= r
    int mid = (lo + hi + 1) >> 1;
    if (cum[mid] <= r) lo = mid; else hi = mid - 1;
  }
  rowmap[r] = lo * bs[lo] + (r - cum[lo]);   // reproduces i*bs[i] + j slice
}

// ---- fused LSTM-cell GEMM (zero state) ------------------------------------
// gates[m, gate*512 + n] = sum_k X[m,k] * W[gate*512+n, k]   (NT gemm)
// epilogue: c = sig(i)*tanh(g); h = sig(o)*tanh(c)
__global__ __launch_bounds__(256)
void lstm_gemm_kernel(const float* __restrict__ X, const int* __restrict__ rowmap,
                      const float* __restrict__ W, const float* __restrict__ b_ih,
                      const float* __restrict__ b_hh,
                      float* __restrict__ Hout, float* __restrict__ Cout,
                      float* __restrict__ Hdup, int Mrows) {
  // LDS tiles stored pre-swizzled in WMMA fragment order:
  //   A: [mtile(8)][lane(32)][elem(16)]  (lane = mrow + 16*khalf)
  //   B: [gate(3)][ntile(2)][lane(32)][elem(16)]
  __shared__ u16 sAhi[BM * BK];
  __shared__ u16 sAlo[BM * BK];
  __shared__ u16 sBhi[NGATE * 32 * BK];
  __shared__ u16 sBlo[NGATE * 32 * BK];

  const int t     = threadIdx.x;
  const int lane  = t & 31;
  const int wave  = t >> 5;
  const int waveM = wave >> 1;   // 0..3  -> 32 rows each
  const int waveN = wave & 1;    // 0..1  -> 16 cols per gate each
  const int m0    = blockIdx.x * BM;
  const int n0    = blockIdx.y * 32;          // per-gate column base (0..511)

  // --- per-thread global-load descriptors (invariant over K loop) ---
  // A rows past Mrows are CLAMPED (not predicated): garbage only reaches
  // C rows >= Mrows, whose stores are guarded in the epilogue.
  const float4* aPtr[4]; int aIdx[4];
#pragma unroll
  for (int c = 0; c < 4; ++c) {
    int vec  = c * 256 + t;          // 0..1023 : covers 128x32 floats as float4
    int mrow = vec >> 3;             // 0..127
    int c4   = vec & 7;
    int k    = c4 * 4;
    int gm   = m0 + mrow;
    if (gm >= Mrows) gm = Mrows - 1;
    int src  = rowmap ? rowmap[gm] : gm;
    aPtr[c]  = (const float4*)(X + (size_t)src * KDIM) + c4;
    int mt = mrow >> 4, mr = mrow & 15;
    int bs = (k >> 3) & 1;                       // lane half select
    int e0 = (k & 7) + 8 * (k >> 4);             // A fragment element index
    aIdx[c] = ((mt * 32 + bs * 16 + mr) << 4) + e0;
  }
  const int gmap[NGATE] = {0, 2, 3};             // i, g, o rows of W
  const float4* bPtr[3]; int bIdx[3];
#pragma unroll
  for (int c = 0; c < 3; ++c) {
    int vec = c * 256 + t;           // 0..767 : covers 96x32 floats as float4
    int wl  = vec >> 3;              // 0..95
    int c4  = vec & 7;
    int k   = c4 * 4;
    int gt  = wl >> 5;               // gate slot 0..2
    int ncl = wl & 31;
    int wr  = gmap[gt] * HDIM + n0 + ncl;
    bPtr[c] = (const float4*)(W + (size_t)wr * KDIM) + c4;
    int ni = ncl >> 4, nr = ncl & 15;
    int bs = (k >> 3) & 1;
    int e0 = (k & 7) + 8 * (k >> 4);             // B fragment element index
    bIdx[c] = (((gt * 2 + ni) * 32 + bs * 16 + nr) << 4) + e0;
  }

  v8f acc[2][NGATE] = {};

  // --- software pipeline: global loads of step kt+1 issue before step kt's
  // WMMA block, so VMEM latency hides behind matrix work.
  float4 av[4], bv[3];
#pragma unroll
  for (int c = 0; c < 4; ++c) av[c] = aPtr[c][0];
#pragma unroll
  for (int c = 0; c < 3; ++c) bv[c] = bPtr[c][0];

  for (int kt = 0; kt < KT; ++kt) {
    __syncthreads();   // previous iteration's fragment reads complete

#pragma unroll
    for (int c = 0; c < 4; ++c) {
      float f[4] = {av[c].x, av[c].y, av[c].z, av[c].w};
      u16x4 hi, lo;
#pragma unroll
      for (int j = 0; j < 4; ++j) {
        u16 h = f2bf(f[j]);
        hi[j] = h;
        lo[j] = f2bf(f[j] - bf2f(h));            // bf16 split: x = hi + lo
      }
      *(u16x4*)(sAhi + aIdx[c]) = hi;
      *(u16x4*)(sAlo + aIdx[c]) = lo;
    }
#pragma unroll
    for (int c = 0; c < 3; ++c) {
      float f[4] = {bv[c].x, bv[c].y, bv[c].z, bv[c].w};
      u16x4 hi, lo;
#pragma unroll
      for (int j = 0; j < 4; ++j) {
        u16 h = f2bf(f[j]);
        hi[j] = h;
        lo[j] = f2bf(f[j] - bf2f(h));
      }
      *(u16x4*)(sBhi + bIdx[c]) = hi;
      *(u16x4*)(sBlo + bIdx[c]) = lo;
    }

    __syncthreads();   // tiles visible

    // prefetch next K-tile (wrapped index: last iteration re-loads tile 0,
    // branch-free and harmless)
    {
      int kn = (kt + 1) & (KT - 1);
#pragma unroll
      for (int c = 0; c < 4; ++c) av[c] = aPtr[c][kn * (BK / 4)];
#pragma unroll
      for (int c = 0; c < 3; ++c) bv[c] = bPtr[c][kn * (BK / 4)];
    }

    BFrag aH[2], aL[2];
#pragma unroll
    for (int mi = 0; mi < 2; ++mi) {
      int base = ((waveM * 2 + mi) * 32 + lane) << 4;
      aH[mi] = ldsFrag(sAhi + base);
      aL[mi] = ldsFrag(sAlo + base);
    }
#pragma unroll
    for (int g = 0; g < NGATE; ++g) {
      int bbase = ((g * 2 + waveN) * 32 + lane) << 4;
      BFrag bH = ldsFrag(sBhi + bbase);
      BFrag bL = ldsFrag(sBlo + bbase);
#pragma unroll
      for (int mi = 0; mi < 2; ++mi) {
        // fp32-fidelity product: hi*hi + hi*lo + lo*hi, fp32 accumulate
        acc[mi][g] = __builtin_amdgcn_wmma_f32_16x16x32_bf16(
            false, aH[mi].v, false, bH.v, (short)0, acc[mi][g], false, false);
        acc[mi][g] = __builtin_amdgcn_wmma_f32_16x16x32_bf16(
            false, aH[mi].v, false, bL.v, (short)0, acc[mi][g], false, false);
        acc[mi][g] = __builtin_amdgcn_wmma_f32_16x16x32_bf16(
            false, aL[mi].v, false, bH.v, (short)0, acc[mi][g], false, false);
      }
    }
  }

  // --- epilogue: bias + LSTM activation, wave-local (same (m,n) across gates)
  const int ncol = n0 + waveN * 16 + (lane & 15);
  float bi = b_ih[ncol]             + b_hh[ncol];
  float bg = b_ih[2 * HDIM + ncol]  + b_hh[2 * HDIM + ncol];
  float bo = b_ih[3 * HDIM + ncol]  + b_hh[3 * HDIM + ncol];
  const int rhalf = (lane >> 4) * 8;    // C layout: VGPR v -> M = v + 8*(lane>=16)
#pragma unroll
  for (int mi = 0; mi < 2; ++mi) {
    int mbase = m0 + waveM * 32 + mi * 16 + rhalf;
#pragma unroll
    for (int v = 0; v < 8; ++v) {
      int m = mbase + v;
      if (m < Mrows) {
        float cc = sigmoidf_(acc[mi][0][v] + bi) * tanhf_(acc[mi][1][v] + bg);
        float hh = sigmoidf_(acc[mi][2][v] + bo) * tanhf_(cc);
        size_t off = (size_t)m * HDIM + ncol;
        Hout[off] = hh;
        Cout[off] = cc;
        if (Hdup) Hdup[off] = hh;     // layer 4 duplicates h4 into `output`
      }
    }
  }
}

extern "C" void kernel_launch(void* const* d_in, const int* in_sizes, int n_in,
                              void* d_out, int out_size, void* d_ws, size_t ws_size,
                              hipStream_t stream) {
  const float* packed_x    = (const float*)d_in[0];
  const int*   batch_sizes = (const int*)d_in[1];
  const int T = in_sizes[1];               // 512
  const int N = in_sizes[0] / KDIM;        // 32643 packed rows

  float* out = (float*)d_out;
  size_t seg = (size_t)N * HDIM;
  float* h1 = out + 1 * seg; float* c1 = out + 2 * seg;
  float* h2 = out + 3 * seg; float* c2 = out + 4 * seg;
  float* h3 = out + 5 * seg; float* c3 = out + 6 * seg;
  float* h4 = out + 7 * seg; float* c4 = out + 8 * seg;

  int* cum    = (int*)d_ws;                // T+1 ints
  int* rowmap = cum + (T + 1);             // N ints

  scan_kernel<<<1, 1, 0, stream>>>(batch_sizes, cum, T);
  rowmap_kernel<<<(N + 255) / 256, 256, 0, stream>>>(batch_sizes, cum, rowmap, N, T);

  dim3 grid((N + BM - 1) / BM, HDIM / 32);
  dim3 blk(256);

  // layer l inputs: W_ih = d_in[2+4(l-1)], b_ih = +2, b_hh = +3 (W_hh unused)
  lstm_gemm_kernel<<<grid, blk, 0, stream>>>(packed_x, rowmap,
      (const float*)d_in[2],  (const float*)d_in[4],  (const float*)d_in[5],
      h1, c1, nullptr, N);
  lstm_gemm_kernel<<<grid, blk, 0, stream>>>(h1, nullptr,
      (const float*)d_in[6],  (const float*)d_in[8],  (const float*)d_in[9],
      h2, c2, nullptr, N);
  lstm_gemm_kernel<<<grid, blk, 0, stream>>>(h2, nullptr,
      (const float*)d_in[10], (const float*)d_in[12], (const float*)d_in[13],
      h3, c3, nullptr, N);
  lstm_gemm_kernel<<<grid, blk, 0, stream>>>(h3, nullptr,
      (const float*)d_in[14], (const float*)d_in[16], (const float*)d_in[17],
      h4, c4, out, N);
}
